// ConvNet_STDP_27951647162588
// MI455X (gfx1250) — matrix operand order repaired and checked by reading, not compiled, not run on hardware
//
#include <hip/hip_runtime.h>

// ---------------------------------------------------------------------------
// CDNA5 (gfx1250) ConvNet+STDP scan.
// All contractions run on V_WMMA_F32_16X16X32_F16. Every WMMA operand is two
// b128 loads: conv B reads n-major patch images, STDP B reads ck-major patch
// images, STDP A reads oc-major f16 spike/trace images, conv A reads LDS
// weight tiles staged by the Tensor Data Mover (tensor_load_to_lds).
// ---------------------------------------------------------------------------

typedef __attribute__((ext_vector_type(16))) _Float16 v16h;
typedef __attribute__((ext_vector_type(8)))  _Float16 v8h;
typedef __attribute__((ext_vector_type(8)))  float    v8f;
typedef __attribute__((ext_vector_type(4)))  unsigned int u32x4;
typedef __attribute__((ext_vector_type(8)))  unsigned int u32x8;

#define BATCH 256
#define SEQ   32
#define L1 102400            // B*400  (conv1 / stdp1 reduce length)
#define L2 9216              // B*36   (conv2 / stdp2 reduce length)

__device__ __forceinline__ v8f wmma16(v16h a, v16h b, v8f c) {
  return __builtin_amdgcn_wmma_f32_16x16x32_f16(false, a, false, b,
                                                (short)0, c, false, false);
}
__device__ __forceinline__ v16h mk16(v8h lo, v8h hi) {
  return __builtin_shufflevector(lo, hi, 0, 1, 2, 3, 4, 5, 6, 7,
                                 8, 9, 10, 11, 12, 13, 14, 15);
}
// A fragment (16x32 f16): row = lane&15; e<8 -> K=akb+e, e>=8 -> K=akb+16+(e-8).
__device__ __forceinline__ v16h loadA(const _Float16* rowp, int akb) {
  v8h lo = *(const v8h*)(rowp + akb);
  v8h hi = *(const v8h*)(rowp + akb + 16);
  return mk16(lo, hi);
}
// B fragment (32x16 f16): col = lane&15; e<8 -> K=kb/2+e, e>=8 -> K=kb/2+16+(e-8).
__device__ __forceinline__ v16h loadB(const _Float16* colp, int kb) {
  v8h lo = *(const v8h*)(colp + kb / 2);
  v8h hi = *(const v8h*)(colp + kb / 2 + 16);
  return mk16(lo, hi);
}
// 1-D Tensor Data Mover copy: global -> LDS, nelems f16 elements (contiguous).
// D# per CDNA5 ISA (group0 + group1, VADDR2/3 = NULL).
__device__ __forceinline__ void tdm_load_to_lds(const _Float16* g,
                                                _Float16* lds, unsigned nelems) {
  unsigned lds_off = (unsigned)(uintptr_t)lds;   // generic low 32 = LDS offset
  unsigned long long ga = (unsigned long long)(uintptr_t)g;
  u32x4 g0;
  g0[0] = 1u;                                               // count = 1
  g0[1] = lds_off;                                          // lds_addr
  g0[2] = (unsigned)ga;                                     // global_addr[31:0]
  g0[3] = (unsigned)((ga >> 32) & 0x01FFFFFFu) | (2u << 30);// addr[56:32], type=2
  u32x8 g1;
  g1[0] = 1u << 16;                                         // data_size = 2 bytes
  g1[1] = (nelems & 0xFFFFu) << 16;                         // tensor_dim0 lo
  g1[2] = nelems >> 16;                                     // tensor_dim0 hi
  g1[3] = (nelems & 0xFFFFu) << 16;                         // tile_dim0 (16-bit)
  g1[4] = 0u;
  g1[5] = nelems;                                           // tensor_dim0_stride
  g1[6] = 0u;
  g1[7] = 0u;
  asm volatile("tensor_load_to_lds %0, %1" :: "s"(g0), "s"(g1) : "memory");
}

// -------------------------------- utility ----------------------------------
__global__ void k_zero(float* __restrict__ p, int n) {
  int i = blockIdx.x * 256 + threadIdx.x;
  if (i < n) p[i] = 0.f;
}
__global__ void k_pack_w1(const float* __restrict__ w, _Float16* __restrict__ wh) {
  int i = blockIdx.x * 256 + threadIdx.x;            // 32 x 64
  if (i >= 32 * 64) return;
  int j = i >> 6, k = i & 63;
  wh[i] = (_Float16)((j < 30 && k < 50) ? w[j * 50 + k] : 0.f);
}
__global__ void k_pack_w2(const float* __restrict__ w, _Float16* __restrict__ wh) {
  int i = blockIdx.x * 256 + threadIdx.x;            // 112 x 768
  if (i >= 112 * 768) return;
  int j = i / 768, k = i % 768;
  wh[i] = (_Float16)((j < 100 && k < 750) ? w[j * 750 + k] : 0.f);
}

// --------------------------- pre-trace 1 update -----------------------------
__global__ void k_trace_pre1(const float* __restrict__ xt, float* __restrict__ tp1) {
  int i = blockIdx.x * 256 + threadIdx.x;            // B*2*576
  if (i < BATCH * 2 * 576) tp1[i] = tp1[i] * 0.98f + xt[i];
}

// -------------- layer-1 patch packs (n-major for conv, k-major for STDP) ----
__device__ __forceinline__ int patch1_src(int n, int k) {
  int b = n / 400, s = n % 400;
  int oy = s / 20, ox = s % 20;
  int c = k / 25, rr = (k % 25) / 5, cc = k % 5;
  return (b * 2 + c) * 576 + (oy + rr) * 24 + (ox + cc);
}
__global__ void k_pack1_nk(const float* __restrict__ xt, _Float16* __restrict__ Px) {
  int i = blockIdx.x * 256 + threadIdx.x;            // L1*64, n-major
  if (i >= L1 * 64) return;
  int n = i >> 6, k = i & 63;
  Px[i] = (_Float16)(k < 50 ? xt[patch1_src(n, k)] : 0.f);
}
__global__ void k_pack1_kn(const float* __restrict__ xt, const float* __restrict__ tp1,
                           _Float16* __restrict__ PxT, _Float16* __restrict__ Ptp1T) {
  int i = blockIdx.x * 256 + threadIdx.x;            // 64*L1, k-major
  if (i >= 64 * L1) return;
  int k = i / L1, n = i % L1;
  float vx = 0.f, vt = 0.f;
  if (k < 50) {
    int idx = patch1_src(n, k);
    vx = xt[idx]; vt = tp1[idx];
  }
  PxT[i] = (_Float16)vx;
  Ptp1T[i] = (_Float16)vt;
}

// ------------------------- conv1 + IAF + post-trace -------------------------
__global__ __launch_bounds__(256) void k_conv1_iaf(
    const _Float16* __restrict__ Px, const _Float16* __restrict__ w1h,
    float* __restrict__ v1, float* __restrict__ to1,
    _Float16* __restrict__ z2h, _Float16* __restrict__ to1h) {
  __shared__ _Float16 wlds[32 * 64];                 // 4 KB via TDM
  int tid = threadIdx.x;
  if (tid < 32) tdm_load_to_lds(w1h, wlds, 32 * 64);
  __builtin_amdgcn_s_wait_tensorcnt(0);
  __syncthreads();

  int lane = tid & 31, wv = tid >> 5;
  int col = blockIdx.x * 128 + wv * 16 + (lane & 15);   // n = b*400 + s
  int b = col / 400, s = col % 400;
  int row = lane & 15, akb = (lane >> 4) * 8, kb = (lane >> 4) * 16;
  const _Float16* colp = Px + (size_t)col * 64;
  __builtin_prefetch(colp, 0, 1);

  v8f acc0 = {}, acc1 = {};
#pragma unroll
  for (int kc = 0; kc < 2; ++kc) {
    v16h bf = loadB(colp + kc * 32, kb);
    v16h a0 = loadA(wlds + row * 64 + kc * 32, akb);
    v16h a1 = loadA(wlds + (row + 16) * 64 + kc * 32, akb);
    acc0 = wmma16(a0, bf, acc0);
    acc1 = wmma16(a1, bf, acc1);
  }
#pragma unroll
  for (int r = 0; r < 8; ++r) {
    int M = r + (lane >> 4) * 8;
    {
      int idx = (b * 30 + M) * 400 + s;
      float vv = v1[idx] + acc0[r];
      float z = vv > 15.0f ? 1.f : 0.f;
      v1[idx] = vv * (1.f - z);
      float tn = to1[idx] * 0.98f + z;
      to1[idx] = tn;
      z2h[M * L1 + col] = (_Float16)z;
      to1h[M * L1 + col] = (_Float16)tn;
    }
    int oc = 16 + M;
    if (oc < 30) {
      int idx = (b * 30 + oc) * 400 + s;
      float vv = v1[idx] + acc1[r];
      float z = vv > 15.0f ? 1.f : 0.f;
      v1[idx] = vv * (1.f - z);
      float tn = to1[idx] * 0.98f + z;
      to1[idx] = tn;
      z2h[oc * L1 + col] = (_Float16)z;
      to1h[oc * L1 + col] = (_Float16)tn;
    }
  }
}

// ------------------------------ STDP layer 1 --------------------------------
// dw+[oc,ck] = sum_l z2h[oc][l] * Ptp1T[ck][l] ; dw-: to1h x PxT.
__global__ __launch_bounds__(256) void k_stdp1(
    const _Float16* __restrict__ PxT, const _Float16* __restrict__ Ptp1T,
    const _Float16* __restrict__ z2h, const _Float16* __restrict__ to1h,
    float* __restrict__ dwp, float* __restrict__ dwm) {
  int tid = threadIdx.x, lane = tid & 31, wv = tid >> 5;
  int mt = wv >> 2, nt = wv & 3;
  int row = lane & 15, akb = (lane >> 4) * 8, kb = (lane >> 4) * 16;
  int oc = mt * 16 + row;                       // rows 30,31 stay zero
  int i = nt * 16 + (lane & 15);                // cols 50..63 stay zero
  const _Float16* ap_row = z2h + (size_t)oc * L1;
  const _Float16* am_row = to1h + (size_t)oc * L1;
  const _Float16* bp_row = Ptp1T + (size_t)i * L1;
  const _Float16* bm_row = PxT + (size_t)i * L1;

  v8f accp = {}, accm = {};
  int Lb0 = blockIdx.x * 1024;
  for (int kc = 0; kc < 32; ++kc) {
    int Lb = Lb0 + kc * 32;
    v16h ap = loadA(ap_row + Lb, akb);
    v16h am = loadA(am_row + Lb, akb);
    v16h bp = loadB(bp_row + Lb, kb);
    v16h bm = loadB(bm_row + Lb, kb);
    accp = wmma16(ap, bp, accp);
    accm = wmma16(am, bm, accm);
  }
#pragma unroll
  for (int r = 0; r < 8; ++r) {
    int M = mt * 16 + r + (lane >> 4) * 8;
    int N = nt * 16 + (lane & 15);
    atomicAdd(&dwp[M * 64 + N], accp[r]);
    atomicAdd(&dwm[M * 64 + N], accm[r]);
  }
}

// --------------------------- weight updates + clip --------------------------
__global__ void k_upd_w1(float* __restrict__ w, float* __restrict__ dwp,
                         float* __restrict__ dwm) {
  int i = blockIdx.x * 256 + threadIdx.x;
  if (i >= 32 * 64) return;
  int j = i >> 6, k = i & 63;
  if (j < 30 && k < 50) {
    float v = w[j * 50 + k] + 0.004f * dwp[i] - 0.003f * dwm[i];
    w[j * 50 + k] = fminf(fmaxf(v, 0.f), 1.f);
  }
  dwp[i] = 0.f; dwm[i] = 0.f;
}
__global__ void k_upd_w2(float* __restrict__ w, float* __restrict__ dwp,
                         float* __restrict__ dwm) {
  int i = blockIdx.x * 256 + threadIdx.x;
  if (i >= 112 * 768) return;
  int j = i / 768, k = i % 768;
  if (j < 100 && k < 750) {
    float v = w[j * 750 + k] + 0.004f * dwp[i] - 0.003f * dwm[i];
    w[j * 750 + k] = fminf(fmaxf(v, 0.f), 1.f);
  }
  dwp[i] = 0.f; dwm[i] = 0.f;
}

// ------------------- maxpool (z2h -> z3h) + pre-trace 2 ---------------------
__global__ void k_pool1(const _Float16* __restrict__ z2h,
                        _Float16* __restrict__ z3h, float* __restrict__ tp2) {
  int i = blockIdx.x * 256 + threadIdx.x;            // (b*30+c)*100 + s
  if (i >= BATCH * 30 * 100) return;
  int bc = i / 100, s = i % 100;
  int b = bc / 30, c = bc % 30;
  int oy = s / 10, ox = s % 10;
  const _Float16* zb = z2h + (size_t)c * L1 + b * 400;
  float m0 = (float)zb[(2 * oy) * 20 + 2 * ox];
  float m1 = (float)zb[(2 * oy) * 20 + 2 * ox + 1];
  float m2 = (float)zb[(2 * oy + 1) * 20 + 2 * ox];
  float m3 = (float)zb[(2 * oy + 1) * 20 + 2 * ox + 1];
  float m = fmaxf(fmaxf(m0, m1), fmaxf(m2, m3));
  z3h[i] = (_Float16)m;
  tp2[i] = tp2[i] * 0.98f + m;
}

// -------------- layer-2 patch packs (n-major for conv, k-major for STDP) ----
__device__ __forceinline__ int patch2_src(int n, int k) {
  int b = n / 36, s = n % 36;
  int oy = s / 6, ox = s % 6;
  int c = k / 25, rr = (k % 25) / 5, cc = k % 5;
  return (b * 30 + c) * 100 + (oy + rr) * 10 + (ox + cc);
}
__global__ void k_pack2_nk(const _Float16* __restrict__ z3h, _Float16* __restrict__ Pz3) {
  int i = blockIdx.x * 256 + threadIdx.x;            // L2*768, n-major
  if (i >= L2 * 768) return;
  int n = i / 768, k = i % 768;
  Pz3[i] = (k < 750) ? z3h[patch2_src(n, k)] : (_Float16)0.f;
}
__global__ void k_pack2_kn(const _Float16* __restrict__ z3h, const float* __restrict__ tp2,
                           _Float16* __restrict__ Pz3T, _Float16* __restrict__ Ptp2T) {
  int i = blockIdx.x * 256 + threadIdx.x;            // 768*L2, k-major
  if (i >= 768 * L2) return;
  int k = i / L2, n = i % L2;
  _Float16 vz = (_Float16)0.f;
  float vt = 0.f;
  if (k < 750) {
    int idx = patch2_src(n, k);
    vz = z3h[idx]; vt = tp2[idx];
  }
  Pz3T[i] = vz;
  Ptp2T[i] = (_Float16)vt;
}

// ------------------------- conv2 (x10) + IAF + trace ------------------------
__global__ __launch_bounds__(256) void k_conv2_iaf(
    const _Float16* __restrict__ Pz3, const _Float16* __restrict__ w2h,
    float* __restrict__ v2, float* __restrict__ to2,
    _Float16* __restrict__ z5h, _Float16* __restrict__ to2h) {
  __shared__ _Float16 wlds[16 * 768];                // 24 KB via TDM
  int tid = threadIdx.x;
  int mt = blockIdx.y;                               // oc tile 0..6
  if (tid < 32) tdm_load_to_lds(w2h + (size_t)mt * 16 * 768, wlds, 16 * 768);
  __builtin_amdgcn_s_wait_tensorcnt(0);
  __syncthreads();

  int lane = tid & 31, wv = tid >> 5;
  int col = blockIdx.x * 128 + wv * 16 + (lane & 15);    // n = b*36 + s
  int b = col / 36, s = col % 36;
  int row = lane & 15, akb = (lane >> 4) * 8, kb = (lane >> 4) * 16;
  const _Float16* colp = Pz3 + (size_t)col * 768;
  __builtin_prefetch(colp, 0, 1);

  v8f acc = {};
  for (int kc = 0; kc < 24; ++kc) {
    v16h bf = loadB(colp + kc * 32, kb);
    v16h a0 = loadA(wlds + row * 768 + kc * 32, akb);
    acc = wmma16(a0, bf, acc);
  }
#pragma unroll
  for (int r = 0; r < 8; ++r) {
    int oc = mt * 16 + r + (lane >> 4) * 8;
    if (oc < 100) {
      int idx = (b * 100 + oc) * 36 + s;
      float vv = v2[idx] + 10.0f * acc[r];
      float z = vv > 10.0f ? 1.f : 0.f;
      v2[idx] = vv * (1.f - z);
      float tn = to2[idx] * 0.98f + z;
      to2[idx] = tn;
      z5h[oc * L2 + col] = (_Float16)z;
      to2h[oc * L2 + col] = (_Float16)tn;
    }
  }
}

// ------------------------------ STDP layer 2 --------------------------------
__global__ __launch_bounds__(256) void k_stdp2(
    const _Float16* __restrict__ Pz3T, const _Float16* __restrict__ Ptp2T,
    const _Float16* __restrict__ z5h, const _Float16* __restrict__ to2h,
    float* __restrict__ dwp, float* __restrict__ dwm) {
  int tid = threadIdx.x, lane = tid & 31, wv = tid >> 5;
  int mt = blockIdx.y, nt = blockIdx.x;
  int row = lane & 15, akb = (lane >> 4) * 8, kb = (lane >> 4) * 16;
  int oc = mt * 16 + row;                       // rows 100..111 stay zero
  int i = nt * 16 + (lane & 15);                // cols 750..767 stay zero
  const _Float16* ap_row = z5h + (size_t)oc * L2;
  const _Float16* am_row = to2h + (size_t)oc * L2;
  const _Float16* bp_row = Ptp2T + (size_t)i * L2;
  const _Float16* bm_row = Pz3T + (size_t)i * L2;

  v8f accp = {}, accm = {};
  for (int kc = wv; kc < 288; kc += 8) {
    int Lb = kc * 32;
    v16h ap = loadA(ap_row + Lb, akb);
    v16h am = loadA(am_row + Lb, akb);
    v16h bp = loadB(bp_row + Lb, kb);
    v16h bm = loadB(bm_row + Lb, kb);
    accp = wmma16(ap, bp, accp);
    accm = wmma16(am, bm, accm);
  }
#pragma unroll
  for (int r = 0; r < 8; ++r) {
    int M = mt * 16 + r + (lane >> 4) * 8;
    int N = nt * 16 + (lane & 15);
    atomicAdd(&dwp[M * 768 + N], accp[r]);
    atomicAdd(&dwm[M * 768 + N], accm[r]);
  }
}

// ------------------------------ global max pool -----------------------------
__global__ void k_z6max(const _Float16* __restrict__ z5h, float* __restrict__ z6) {
  int i = blockIdx.x * 256 + threadIdx.x;            // b*100 + oc
  if (i >= BATCH * 100) return;
  int b = i / 100, oc = i % 100;
  const _Float16* p = z5h + (size_t)oc * L2 + b * 36;
  float m = (float)p[0];
  for (int j = 1; j < 36; ++j) m = fmaxf(m, (float)p[j]);
  z6[i] = m;
}

// ----------------------- readout: fc1 + relu + LI cell ----------------------
__global__ __launch_bounds__(64) void k_readout(
    const float* __restrict__ z6, const float* __restrict__ fc1_w,
    const float* __restrict__ fc1_b, const float* __restrict__ out_w,
    float* __restrict__ li_v, float* __restrict__ li_i,
    float* __restrict__ out_t) {
  __shared__ float zs[100];
  __shared__ float hs[50];
  int b = blockIdx.x, tid = threadIdx.x;
  for (int k = tid; k < 100; k += 64) zs[k] = z6[b * 100 + k];
  __syncthreads();
  if (tid < 50) {
    float a = fc1_b[tid];
    for (int k = 0; k < 100; ++k) a += zs[k] * fc1_w[tid * 100 + k];
    hs[tid] = a > 0.f ? a : 0.f;
  }
  __syncthreads();
  if (tid < 10) {
    float vo = li_v[b * 10 + tid], io = li_i[b * 10 + tid];
    float vn = vo + 0.1f * (io - vo);
    float s = 0.f;
    for (int k = 0; k < 50; ++k) s += hs[k] * out_w[tid * 50 + k];
    li_i[b * 10 + tid] = 0.8f * io + s;
    li_v[b * 10 + tid] = vn;
    out_t[b * 10 + tid] = vn;
  }
}

// ------------------------------- host launcher ------------------------------
extern "C" void kernel_launch(void* const* d_in, const int* in_sizes, int n_in,
                              void* d_out, int out_size, void* d_ws, size_t ws_size,
                              hipStream_t stream) {
  const float* x      = (const float*)d_in[0];
  const float* w1_in  = (const float*)d_in[1];
  const float* w2_in  = (const float*)d_in[2];
  const float* fc1_w  = (const float*)d_in[3];
  const float* fc1_b  = (const float*)d_in[4];
  const float* out_w  = (const float*)d_in[5];
  float* out = (float*)d_out;

  char* base = (char*)d_ws;
  size_t off = 0;
  auto takef = [&](size_t n) -> float* {
    float* p = (float*)(base + off);
    off += ((n * sizeof(float) + 255) & ~(size_t)255);
    return p;
  };
  auto takeh = [&](size_t n) -> _Float16* {
    _Float16* p = (_Float16*)(base + off);
    off += ((n * sizeof(_Float16) + 255) & ~(size_t)255);
    return p;
  };
  // fp32 state
  float* v1   = takef(3072000);      // B*30*400
  float* v2   = takef(921600);       // B*100*36
  float* tp1  = takef(294912);       // B*2*576
  float* to1  = takef(3072000);
  float* tp2  = takef(768000);       // B*30*100
  float* to2  = takef(921600);
  float* z6   = takef(25600);
  float* li_v = takef(2560);
  float* li_i = takef(2560);
  float* w1f  = takef(1500);
  float* w2f  = takef(75000);
  float* dwp1 = takef(2048);
  float* dwm1 = takef(2048);
  float* dwp2 = takef(86016);
  float* dwm2 = takef(86016);
  // f16 images (pad rows/cols stay zero: zeroed at init, never written)
  _Float16* w1h   = takeh(32 * 64);
  _Float16* w2h   = takeh(112 * 768);
  _Float16* Px    = takeh((size_t)L1 * 64);    // n-major (conv1 B)
  _Float16* PxT   = takeh((size_t)64 * L1);    // k-major (stdp1 B-)
  _Float16* Ptp1T = takeh((size_t)64 * L1);    // k-major (stdp1 B+)
  _Float16* Pz3   = takeh((size_t)L2 * 768);   // n-major (conv2 B)
  _Float16* Pz3T  = takeh((size_t)768 * L2);   // k-major (stdp2 B-)
  _Float16* Ptp2T = takeh((size_t)768 * L2);   // k-major (stdp2 B+)
  _Float16* z2h   = takeh((size_t)32 * L1);
  _Float16* to1h  = takeh((size_t)32 * L1);
  _Float16* z3h   = takeh(768000);
  _Float16* z5h   = takeh((size_t)112 * L2);
  _Float16* to2h  = takeh((size_t)112 * L2);
  size_t nzero = off / sizeof(float);

  k_zero<<<(unsigned)((nzero + 255) / 256), 256, 0, stream>>>((float*)d_ws, (int)nzero);
  hipMemcpyAsync(w1f, w1_in, 1500 * sizeof(float), hipMemcpyDeviceToDevice, stream);
  hipMemcpyAsync(w2f, w2_in, 75000 * sizeof(float), hipMemcpyDeviceToDevice, stream);

  for (int t = 0; t < SEQ; ++t) {
    const float* xt = x + (size_t)t * BATCH * 2 * 576;
    // ----- layer 1 -----
    k_trace_pre1<<<1152, 256, 0, stream>>>(xt, tp1);
    k_pack1_nk<<<25600, 256, 0, stream>>>(xt, Px);
    k_pack1_kn<<<25600, 256, 0, stream>>>(xt, tp1, PxT, Ptp1T);
    k_pack_w1<<<8, 256, 0, stream>>>(w1f, w1h);
    k_conv1_iaf<<<800, 256, 0, stream>>>(Px, w1h, v1, to1, z2h, to1h);
    k_stdp1<<<100, 256, 0, stream>>>(PxT, Ptp1T, z2h, to1h, dwp1, dwm1);
    k_upd_w1<<<8, 256, 0, stream>>>(w1f, dwp1, dwm1);
    // ----- pool + layer 2 -----
    k_pool1<<<3000, 256, 0, stream>>>(z2h, z3h, tp2);
    k_pack2_nk<<<27648, 256, 0, stream>>>(z3h, Pz3);
    k_pack2_kn<<<27648, 256, 0, stream>>>(z3h, tp2, Pz3T, Ptp2T);
    k_pack_w2<<<336, 256, 0, stream>>>(w2f, w2h);
    k_conv2_iaf<<<dim3(72, 7), 256, 0, stream>>>(Pz3, w2h, v2, to2, z5h, to2h);
    k_stdp2<<<dim3(47, 7), 256, 0, stream>>>(Pz3T, Ptp2T, z5h, to2h, dwp2, dwm2);
    k_upd_w2<<<336, 256, 0, stream>>>(w2f, dwp2, dwm2);
    // ----- readout -----
    k_z6max<<<100, 256, 0, stream>>>(z5h, z6);
    k_readout<<<BATCH, 64, 0, stream>>>(z6, fc1_w, fc1_b, out_w, li_v, li_i,
                                        out + (size_t)t * BATCH * 10);
  }
}
